// SlotAttention_78675210928337
// MI455X (gfx1250) — matrix operand order repaired and checked by reading, not compiled, not run on hardware
//
#include <hip/hip_runtime.h>

#define BATCH 64
#define NPTS  16384
#define DIM   64
#define SLOTS 8
#define HID   128
#define NT    256            // rows per block in attention kernel
#define NBLK  (NPTS / NT)    // 64
#define KVT   256            // rows per block in kv kernel

typedef __attribute__((ext_vector_type(16))) __bf16 v16bf;
typedef __attribute__((ext_vector_type(8)))  float  v8f;
typedef __attribute__((ext_vector_type(4)))  int    v4i;

#if defined(__has_builtin)
#if __has_builtin(__builtin_amdgcn_global_load_async_to_lds_b128)
#define HAVE_ASYNC_LDS 1
#endif
#if __has_builtin(__builtin_amdgcn_s_wait_asynccnt)
#define HAVE_WAIT_ASYNC 1
#endif
#endif

__device__ __forceinline__ unsigned short f2bf_us(float f) {
    unsigned u = __builtin_bit_cast(unsigned, f);
    u += 0x7FFFu + ((u >> 16) & 1u);           // round-to-nearest-even
    return (unsigned short)(u >> 16);
}
__device__ __forceinline__ __bf16 us2bf(unsigned short u) {
    return __builtin_bit_cast(__bf16, u);
}

// ---------------------------------------------------------------------------
// slots = mu + exp(log_sigma) * slots_init
// ---------------------------------------------------------------------------
__global__ void init_slots_kernel(const float* __restrict__ init,
                                  const float* __restrict__ mu,
                                  const float* __restrict__ ls,
                                  float* __restrict__ slotsW) {
    int i = blockIdx.x * 256 + threadIdx.x;
    if (i < BATCH * SLOTS * DIM) {
        int d = i & (DIM - 1);
        slotsW[i] = mu[d] + __expf(ls[d]) * init[i];
    }
}

// ---------------------------------------------------------------------------
// x = LN(inputs); k = 0.125 * x @ Wk^T (bf16); v = x @ Wv^T (bf16)
// ---------------------------------------------------------------------------
__global__ __launch_bounds__(256) void kv_kernel(
    const float* __restrict__ inputs,
    const float* __restrict__ g_in, const float* __restrict__ b_in,
    const float* __restrict__ Wk,  const float* __restrict__ Wv,
    unsigned short* __restrict__ kOut, unsigned short* __restrict__ vOut) {
    __shared__ __align__(16) unsigned short xn[KVT * DIM];   // 32 KB
    __shared__ __align__(16) unsigned short wk[DIM * DIM];   // 8 KB
    __shared__ __align__(16) unsigned short wv[DIM * DIM];   // 8 KB

    const int t = threadIdx.x;
    const int lane = t & 31, wave = t >> 5;
    const int g = lane >> 4, ls = lane & 15;
    const int nb = NPTS / KVT;
    const int b  = blockIdx.x / nb;
    const int n0 = (blockIdx.x % nb) * KVT;

    // ---- LayerNorm, one row per thread, write bf16 to LDS
    {
        const float* ip = inputs + ((size_t)b * NPTS + n0 + t) * DIM;
        float m = 0.f;
        for (int c = 0; c < DIM; ++c) m += ip[c];
        m *= (1.0f / DIM);
        float var = 0.f;
        for (int c = 0; c < DIM; ++c) { float d = ip[c] - m; var += d * d; }
        var *= (1.0f / DIM);
        float inv = rsqrtf(var + 1e-5f);
        for (int c = 0; c < DIM; ++c)
            xn[t * DIM + c] = f2bf_us((ip[c] - m) * inv * g_in[c] + b_in[c]);
    }
    for (int i = t; i < DIM * DIM; i += 256) {
        wk[i] = f2bf_us(Wk[i]);
        wv[i] = f2bf_us(Wv[i]);
    }
    __syncthreads();

    // ---- 16 m-tiles x 4 d-tiles, 8 per wave, each k and v
    for (int i = 0; i < 8; ++i) {
        const int T  = wave * 8 + i;
        const int mt = T >> 2, dt = T & 3;
        v16bf af[2];
#pragma unroll
        for (int kk = 0; kk < 2; ++kk) {
            const unsigned short* rp = &xn[(mt * 16 + ls) * DIM + 32 * kk + 16 * g];
#pragma unroll
            for (int e = 0; e < 8; ++e) { af[kk][e] = us2bf(rp[e]); af[kk][8 + e] = us2bf(rp[8 + e]); }
        }
        const int dcol = dt * 16 + ls;
        v8f ak = {}, av = {};
#pragma unroll
        for (int kk = 0; kk < 2; ++kk) {
            v16bf bk, bv;
            const unsigned short* wkp = &wk[dcol * DIM + 32 * kk + 16 * g];
            const unsigned short* wvp = &wv[dcol * DIM + 32 * kk + 16 * g];
#pragma unroll
            for (int e = 0; e < 16; ++e) { bk[e] = us2bf(wkp[e]); bv[e] = us2bf(wvp[e]); }
            ak = __builtin_amdgcn_wmma_f32_16x16x32_bf16(false, af[kk], false, bk, (short)0, ak, false, false);
            av = __builtin_amdgcn_wmma_f32_16x16x32_bf16(false, af[kk], false, bv, (short)0, av, false, false);
        }
#pragma unroll
        for (int j = 0; j < 8; ++j) {
            int n = n0 + mt * 16 + j + 8 * g;
            size_t oi = ((size_t)b * NPTS + n) * DIM + dcol;
            kOut[oi] = f2bf_us(ak[j] * 0.125f);   // fold scale = D^-0.5 into k
            vOut[oi] = f2bf_us(av[j]);
        }
    }
}

// ---------------------------------------------------------------------------
// q = LN(slots) @ Wq^T  (stored bf16, padded to 16 slot rows with zeros)
// ---------------------------------------------------------------------------
__global__ __launch_bounds__(64) void q_kernel(
    const float* __restrict__ slots, const float* __restrict__ Wq,
    const float* __restrict__ g_s, const float* __restrict__ b_s,
    unsigned short* __restrict__ qOut) {
    __shared__ float sn[SLOTS * DIM];
    const int t = threadIdx.x;
    const int b = blockIdx.x;
    if (t < SLOTS) {
        const float* sp = slots + (size_t)b * SLOTS * DIM + t * DIM;
        float m = 0.f;
        for (int c = 0; c < DIM; ++c) m += sp[c];
        m *= (1.0f / DIM);
        float var = 0.f;
        for (int c = 0; c < DIM; ++c) { float d = sp[c] - m; var += d * d; }
        var *= (1.0f / DIM);
        float inv = rsqrtf(var + 1e-5f);
        for (int c = 0; c < DIM; ++c) sn[t * DIM + c] = (sp[c] - m) * inv * g_s[c] + b_s[c];
    }
    __syncthreads();
    const int d = t;
    for (int s = 0; s < SLOTS; ++s) {
        float acc = 0.f;
        for (int c = 0; c < DIM; ++c) acc += sn[s * DIM + c] * Wq[d * DIM + c];
        qOut[((size_t)b * 16 + s) * DIM + d] = f2bf_us(acc);
    }
    for (int s = SLOTS; s < 16; ++s) qOut[((size_t)b * 16 + s) * DIM + d] = 0;
}

// ---------------------------------------------------------------------------
// Fused: logits = k@q^T (WMMA) -> softmax over slots -> accumulate attns/3
// -> a = attn + eps -> partial colsum and partial U = a^T @ v (WMMA).
// v tile is streamed into LDS with async global->LDS copies (ASYNCcnt),
// overlapped with the logits/softmax phase.
// ---------------------------------------------------------------------------
__global__ __launch_bounds__(256) void attn_kernel(
    const unsigned short* __restrict__ kW,
    const unsigned short* __restrict__ vW,
    const unsigned short* __restrict__ qW,
    float* __restrict__ outAll,        // [slots | attns]
    float* __restrict__ Upart,         // [B][NBLK][8][64]
    float* __restrict__ Cpart,         // [B][NBLK][8]
    int it) {
    __shared__ __align__(16) unsigned short qs[16 * DIM];     // 2 KB
    __shared__ __align__(16) unsigned short aT[16 * NT];      // 8 KB (rows 8..15 unused)
    __shared__ __align__(16) unsigned short vTile[NT * DIM];  // 32 KB
    __shared__ float red[4 * 4 * 16 * 16];                    // 16 KB (4-wave K-split)
    __shared__ float wcol[8][8];

    const int t = threadIdx.x;
    const int lane = t & 31, wave = t >> 5;
    const int g = lane >> 4, ls = lane & 15;
    const int blk = blockIdx.x & (NBLK - 1);
    const int b   = blockIdx.x / NBLK;
    const int n0  = blk * NT;

    for (int i = t; i < 16 * DIM; i += 256) qs[i] = qW[(size_t)b * 16 * DIM + i];
    __syncthreads();

    const unsigned short* kb = kW + ((size_t)b * NPTS + n0) * DIM;
    const unsigned short* vb = vW + ((size_t)b * NPTS + n0) * DIM;
    float* attnOut = outAll + BATCH * SLOTS * DIM;

    // ---- kick off async v-tile staging: NT*DIM*2 B = 32 KB = 2048 x 16B
#ifdef HAVE_ASYNC_LDS
#pragma unroll
    for (int i = 0; i < 8; ++i) {
        int chunk = t + i * 256;
        __attribute__((address_space(1))) v4i* gsrc =
            (__attribute__((address_space(1))) v4i*)(vb + chunk * 8);
        __attribute__((address_space(3))) v4i* ldst =
            (__attribute__((address_space(3))) v4i*)(&vTile[chunk * 8]);
        __builtin_amdgcn_global_load_async_to_lds_b128(gsrc, ldst, 0, 0);
    }
#else
    {
        const uint4* s4 = (const uint4*)vb;
        uint4* d4 = (uint4*)vTile;
        for (int i = t; i < (NT * DIM * 2) / 16; i += 256) d4[i] = s4[i];
    }
#endif

    // ---- Phase B: logits + softmax + attn accumulate + aT staging
    float csum = 0.0f;
    for (int i = 0; i < 2; ++i) {
        const int mt = wave * 2 + i;
        v8f acc = {};
#pragma unroll
        for (int kk = 0; kk < 2; ++kk) {
            v16bf af, bf;
            const unsigned short* rp = kb + (size_t)(mt * 16 + ls) * DIM + 32 * kk + 16 * g;
#pragma unroll
            for (int e = 0; e < 8; ++e) { af[e] = us2bf(rp[e]); af[8 + e] = us2bf(rp[8 + e]); }
            const unsigned short* qp = &qs[ls * DIM + 32 * kk + 16 * g];
#pragma unroll
            for (int e = 0; e < 16; ++e) bf[e] = us2bf(qp[e]);
            acc = __builtin_amdgcn_wmma_f32_16x16x32_bf16(false, af, false, bf, (short)0, acc, false, false);
        }
        // softmax over the 8 slot lanes (xor 1,2,4 stays inside the s-group)
#pragma unroll
        for (int j = 0; j < 8; ++j) {
            float val = acc[j];
            float m = val;
            m = fmaxf(m, __shfl_xor(m, 1));
            m = fmaxf(m, __shfl_xor(m, 2));
            m = fmaxf(m, __shfl_xor(m, 4));
            float e = __expf(val - m);
            float sum = e;
            sum += __shfl_xor(sum, 1);
            sum += __shfl_xor(sum, 2);
            sum += __shfl_xor(sum, 4);
            float attn = e / sum;
            int nloc = mt * 16 + j + 8 * g;
            if (ls < 8) {
                size_t oi = ((size_t)b * NPTS + n0 + nloc) * 8 + ls;
                float a3 = attn * (1.0f / 3.0f);
                if (it == 0) attnOut[oi] = a3;
                else         attnOut[oi] += a3;
                float a = attn + 1e-8f;
                aT[ls * NT + nloc] = f2bf_us(a);
                csum += a;
            }
        }
    }
    csum += __shfl_xor(csum, 16);
    if (lane < 8) wcol[wave][lane] = csum;

    // ---- wait for async v staging, then sync all waves
#ifdef HAVE_ASYNC_LDS
#ifdef HAVE_WAIT_ASYNC
    __builtin_amdgcn_s_wait_asynccnt(0);
#else
    asm volatile("s_wait_asynccnt 0" ::: "memory");
#endif
#endif
    __syncthreads();

    // ---- Phase C: U = a^T @ v ; waves 0..3 own K range [w*64, w*64+64)
    if (wave < 4) {
        const int kbase = wave * 64;
        v8f u[4] = {};
#pragma unroll
        for (int kk = 0; kk < 2; ++kk) {
            const int kb0 = kbase + 32 * kk;
            v16bf af;
            const unsigned short* ap = &aT[ls * NT + kb0 + 16 * g];
#pragma unroll
            for (int e = 0; e < 8; ++e) { af[e] = us2bf(ap[e]); af[8 + e] = us2bf(ap[8 + e]); }
#pragma unroll
            for (int nt = 0; nt < 4; ++nt) {
                v16bf bf;
                const int dcol = nt * 16 + ls;
                const unsigned short* vp = &vTile[(kb0 + 16 * g) * DIM + dcol];
#pragma unroll
                for (int e = 0; e < 16; ++e) bf[e] = us2bf(vp[e * DIM]);
                u[nt] = __builtin_amdgcn_wmma_f32_16x16x32_bf16(false, af, false, bf, (short)0, u[nt], false, false);
            }
        }
#pragma unroll
        for (int nt = 0; nt < 4; ++nt)
#pragma unroll
            for (int j = 0; j < 8; ++j)
                red[((wave * 4 + nt) * 16 + (j + 8 * g)) * 16 + ls] = u[nt][j];
    }
    __syncthreads();

    // deterministic cross-wave reduction of the K-split partials
    for (int o = t; o < SLOTS * DIM; o += 256) {
        int s = o >> 6, d = o & 63;
        int nt = d >> 4, col = d & 15;
        float acc = 0.f;
#pragma unroll
        for (int w = 0; w < 4; ++w) acc += red[((w * 4 + nt) * 16 + s) * 16 + col];
        Upart[(((size_t)b * NBLK + blk) * SLOTS + s) * DIM + d] = acc;
    }
    if (t < 8) {
        float acc = 0.f;
#pragma unroll
        for (int w = 0; w < 8; ++w) acc += wcol[w][t];
        Cpart[((size_t)b * NBLK + blk) * SLOTS + t] = acc;
    }
}

// ---------------------------------------------------------------------------
// Reduce partials, normalize updates, GRU cell, LN, MLP, new slots.
// ---------------------------------------------------------------------------
__global__ __launch_bounds__(256) void slot_update_kernel(
    const float* __restrict__ Upart, const float* __restrict__ Cpart,
    float* __restrict__ slotsW,
    const float* __restrict__ w_ih, const float* __restrict__ w_hh,
    const float* __restrict__ b_ih, const float* __restrict__ b_hh,
    const float* __restrict__ g_m, const float* __restrict__ b_m,
    const float* __restrict__ W1, const float* __restrict__ b1,
    const float* __restrict__ W2, const float* __restrict__ b2,
    float* __restrict__ outSlots, int last) {
    __shared__ float upd[512], prev[512], gi[1536], gh[1536];
    __shared__ float hb[512], hn[512], hid[1024], col[8];
    const int t = threadIdx.x;
    const int b = blockIdx.x;

    for (int o = t; o < 512; o += 256) {
        float acc = 0.f;
        for (int blk = 0; blk < NBLK; ++blk)
            acc += Upart[(((size_t)b * NBLK + blk) * SLOTS + (o >> 6)) * DIM + (o & 63)];
        upd[o]  = acc;
        prev[o] = slotsW[(size_t)b * 512 + o];
    }
    if (t < 8) {
        float acc = 0.f;
        for (int blk = 0; blk < NBLK; ++blk) acc += Cpart[((size_t)b * NBLK + blk) * SLOTS + t];
        col[t] = acc;
    }
    __syncthreads();
    for (int o = t; o < 512; o += 256) upd[o] /= col[o >> 6];
    __syncthreads();

    for (int o = t; o < 1536; o += 256) {
        int s = o / 192, kk = o % 192;
        float a1 = b_ih[kk], a2 = b_hh[kk];
        for (int c = 0; c < DIM; ++c) {
            a1 += upd[s * DIM + c]  * w_ih[kk * DIM + c];
            a2 += prev[s * DIM + c] * w_hh[kk * DIM + c];
        }
        gi[o] = a1;
        gh[o] = a2;
    }
    __syncthreads();
    for (int o = t; o < 512; o += 256) {
        int s = o >> 6, j = o & 63;
        float r  = 1.f / (1.f + __expf(-(gi[s * 192 + j] + gh[s * 192 + j])));
        float z  = 1.f / (1.f + __expf(-(gi[s * 192 + 64 + j] + gh[s * 192 + 64 + j])));
        float nn = tanhf(gi[s * 192 + 128 + j] + r * gh[s * 192 + 128 + j]);
        hb[o] = (1.f - z) * nn + z * prev[o];
    }
    __syncthreads();
    if (t < 8) {
        float m = 0.f;
        for (int c = 0; c < DIM; ++c) m += hb[t * DIM + c];
        m *= (1.0f / DIM);
        float var = 0.f;
        for (int c = 0; c < DIM; ++c) { float d = hb[t * DIM + c] - m; var += d * d; }
        var *= (1.0f / DIM);
        float inv = rsqrtf(var + 1e-5f);
        for (int c = 0; c < DIM; ++c) hn[t * DIM + c] = (hb[t * DIM + c] - m) * inv * g_m[c] + b_m[c];
    }
    __syncthreads();
    for (int o = t; o < SLOTS * HID; o += 256) {
        int s = o >> 7, kk = o & 127;
        float a = b1[kk];
        for (int c = 0; c < DIM; ++c) a += hn[s * DIM + c] * W1[kk * DIM + c];
        hid[o] = fmaxf(a, 0.f);
    }
    __syncthreads();
    for (int o = t; o < 512; o += 256) {
        int s = o >> 6, j = o & 63;
        float a = b2[j];
        for (int c = 0; c < HID; ++c) a += hid[s * HID + c] * W2[j * HID + c];
        float ns = hb[o] + a;
        slotsW[(size_t)b * 512 + o] = ns;
        if (last) outSlots[(size_t)b * 512 + o] = ns;
    }
}

// ---------------------------------------------------------------------------
extern "C" void kernel_launch(void* const* d_in, const int* in_sizes, int n_in,
                              void* d_out, int out_size, void* d_ws, size_t ws_size,
                              hipStream_t stream) {
    (void)in_sizes; (void)n_in; (void)out_size; (void)ws_size;
    const float* inputs     = (const float*)d_in[0];
    const float* slots_init = (const float*)d_in[1];
    const float* ln_in_g = (const float*)d_in[2];
    const float* ln_in_b = (const float*)d_in[3];
    const float* ln_s_g  = (const float*)d_in[4];
    const float* ln_s_b  = (const float*)d_in[5];
    const float* ln_m_g  = (const float*)d_in[6];
    const float* ln_m_b  = (const float*)d_in[7];
    const float* Wq      = (const float*)d_in[8];
    const float* Wk      = (const float*)d_in[9];
    const float* Wv      = (const float*)d_in[10];
    const float* w_ih    = (const float*)d_in[11];
    const float* w_hh    = (const float*)d_in[12];
    const float* b_ih    = (const float*)d_in[13];
    const float* b_hh    = (const float*)d_in[14];
    const float* W1      = (const float*)d_in[15];
    const float* b1      = (const float*)d_in[16];
    const float* W2      = (const float*)d_in[17];
    const float* b2      = (const float*)d_in[18];
    const float* slots_mu = (const float*)d_in[19];
    const float* slots_ls = (const float*)d_in[20];

    const size_t kvElems = (size_t)BATCH * NPTS * DIM;
    unsigned short* kW = (unsigned short*)d_ws;
    unsigned short* vW = kW + kvElems;
    float* slotsW = (float*)(vW + kvElems);
    unsigned short* qW = (unsigned short*)(slotsW + (size_t)BATCH * SLOTS * DIM);
    float* UpartW = (float*)(qW + (size_t)BATCH * 16 * DIM);
    float* CpartW = UpartW + (size_t)BATCH * NBLK * SLOTS * DIM;

    float* outF = (float*)d_out;

    init_slots_kernel<<<(BATCH * SLOTS * DIM + 255) / 256, 256, 0, stream>>>(
        slots_init, slots_mu, slots_ls, slotsW);
    kv_kernel<<<BATCH * (NPTS / KVT), 256, 0, stream>>>(
        inputs, ln_in_g, ln_in_b, Wk, Wv, kW, vW);

    for (int it = 0; it < 3; ++it) {
        q_kernel<<<BATCH, 64, 0, stream>>>(slotsW, Wq, ln_s_g, ln_s_b, qW);
        attn_kernel<<<BATCH * NBLK, 256, 0, stream>>>(kW, vW, qW, outF, UpartW, CpartW, it);
        slot_update_kernel<<<BATCH, 256, 0, stream>>>(
            UpartW, CpartW, slotsW, w_ih, w_hh, b_ih, b_hh,
            ln_m_g, ln_m_b, W1, b1, W2, b2, outF, it == 2);
    }
}
